// rgcn_21337397526985
// MI455X (gfx1250) — matrix operand order repaired and checked by reading, not compiled, not run on hardware
//
#include <hip/hip_runtime.h>
#include <hip/hip_bf16.h>
#include <math.h>

// ---------------- types for WMMA fragments ----------------
typedef __attribute__((ext_vector_type(16))) __bf16 v16bf;
typedef __attribute__((ext_vector_type(8)))  __bf16 v8bf;
typedef __attribute__((ext_vector_type(8)))  float  v8f;

#define NN     4096
#define TT     365
#define DD     20
#define HH     32
#define KSPLIT 4

__device__ __forceinline__ unsigned short f2bf(float f) {
  unsigned int u = __float_as_uint(f);
  unsigned int r = (u + 0x7FFFu + ((u >> 16) & 1u)) >> 16;   // round-to-nearest-even
  return (unsigned short)r;
}
__device__ __forceinline__ float sigf(float x) { return 1.0f / (1.0f + expf(-x)); }

__device__ __forceinline__ v16bf mk16(v8bf lo, v8bf hi) {
  union { v16bf v; v8bf h[2]; } u;
  u.h[0] = lo; u.h[1] = hi;
  return u.v;
}

// ---------------- one-time: A fp32 -> bf16 ----------------
__global__ void k_f32_to_bf16(const float* __restrict__ src,
                              unsigned short* __restrict__ dst, int n) {
  int i = blockIdx.x * blockDim.x + threadIdx.x;
  if (i < n) dst[i] = f2bf(src[i]);
}

// ---------------- shared per-node LSTM + graph pre-mix math ----------------
// one wave32 per node; lane == hidden index. hp/cp are the recurrent state.
__device__ __forceinline__ void premix_core(
    float hp, float cp, int lane, int node, int t,
    const float* __restrict__ inputs,
    const float* __restrict__ Wgh, const float* __restrict__ bgh,
    const float* __restrict__ Wgc, const float* __restrict__ bgc,
    const float* __restrict__ Kz,  const float* __restrict__ Rz,
    const float* __restrict__ bz,
    unsigned short* __restrict__ G, float* __restrict__ hcur,
    float* __restrict__ ccur)
{
  const float xv = (lane < DD) ? inputs[(size_t)node * TT * DD + (size_t)t * DD + lane] : 0.0f;

  float gh = bgh[lane], gc = bgc[lane];
  float z0 = bz[lane], z1 = bz[HH + lane], z2 = bz[2 * HH + lane], z3 = bz[3 * HH + lane];

  #pragma unroll 4
  for (int k = 0; k < HH; ++k) {
    const float hpk = __shfl(hp, k, 32);
    const float cpk = __shfl(cp, k, 32);
    gh += hpk * Wgh[k * HH + lane];
    gc += cpk * Wgc[k * HH + lane];
    const float* r = Rz + k * (4 * HH);
    z0 += hpk * r[lane];
    z1 += hpk * r[HH + lane];
    z2 += hpk * r[2 * HH + lane];
    z3 += hpk * r[3 * HH + lane];
  }
  #pragma unroll 4
  for (int k = 0; k < DD; ++k) {
    const float xk = __shfl(xv, k, 32);
    const float* kk = Kz + k * (4 * HH);
    z0 += xk * kk[lane];
    z1 += xk * kk[HH + lane];
    z2 += xk * kk[2 * HH + lane];
    z3 += xk * kk[3 * HH + lane];
  }

  const float ig = sigf(z0), fg = sigf(z1), gg = tanhf(z2), og = sigf(z3);
  const float cc = fg * cp + ig * gg;
  const float hc = og * tanhf(cc);

  ccur[node * HH + lane] = cc;
  hcur[node * HH + lane] = hc;
  G[(size_t)lane * NN + node]        = f2bf(gh);   // G stored [64][NN]; rows 0..31  : h pre-mix
  G[(size_t)(32 + lane) * NN + node] = f2bf(gc);   // rows 32..63 : c pre-mix
}

// t = 0 bootstrap: h_prev = c_prev = 0
__global__ __launch_bounds__(256) void k_premix_first(
    const float* __restrict__ inputs,
    const float* __restrict__ Wgh, const float* __restrict__ bgh,
    const float* __restrict__ Wgc, const float* __restrict__ bgc,
    const float* __restrict__ Kz,  const float* __restrict__ Rz,
    const float* __restrict__ bz,
    unsigned short* __restrict__ G, float* __restrict__ hcur,
    float* __restrict__ ccur)
{
  const int lane = threadIdx.x & 31;
  const int node = blockIdx.x * 8 + (threadIdx.x >> 5);
  premix_core(0.0f, 0.0f, lane, node, 0, inputs, Wgh, bgh, Wgc, bgc, Kz, Rz, bz,
              G, hcur, ccur);
}

// ---------------- per-step: AGp[part] = A_bf16[:, Krange] @ G[Krange, :] ----------------
// wave per 16-row strip x all 64 columns; gridDim.y = KSPLIT for 4x resident waves.
// Explicitly software-pipelined: next chunk's 10 loads are issued before the
// current chunk's 4 WMMAs, with double-buffered fragment registers so the
// scheduler cannot collapse them into a single recycled set (the round-2 binary
// showed s_wait_loadcnt 0x0 before every WMMA because of register reuse).
__global__ __launch_bounds__(128) void k_agemm(
    const unsigned short* __restrict__ Abf,   // [4096,4096] bf16
    const unsigned short* __restrict__ G,     // [4096,64]   bf16
    float* __restrict__ AGp)                  // KSPLIT x [4096,64] f32 partials
{
  const int lane = threadIdx.x & 31;
  const int wave = threadIdx.x >> 5;
  const int row0 = blockIdx.x * 64 + wave * 16;
  const int kbeg = blockIdx.y * (NN / KSPLIT);
  float* AG = AGp + (size_t)blockIdx.y * NN * 64;

  // A fragment per ISA 16-bit A 16x32 layout:
  // lanes 0-15: M=lane, K=[0..7] then [16..23]; lanes 16-31: M=lane-16, K=[8..15] then [24..31]
  const int mrow = row0 + (lane & 15);
  const int asel = (lane >> 4) << 3;            // 0 or 8
  const unsigned short* aptr = Abf + (size_t)mrow * NN + kbeg + asel;
  // B fragment: lane holds column N=lane&15, K in same order as A (G stored [64][NN])
  const unsigned short* gptr = G + (size_t)(lane & 15) * NN + kbeg + asel;

  v8f acc0 = {}, acc1 = {}, acc2 = {}, acc3 = {};

  // prologue: chunk 0
  v8bf cA0 = *(const v8bf*)(aptr);
  v8bf cA1 = *(const v8bf*)(aptr + 16);
  v8bf cB0 = *(const v8bf*)(gptr +  0), cB1 = *(const v8bf*)(gptr + 16);
  v8bf cB2 = *(const v8bf*)(gptr + (size_t)16 * NN), cB3 = *(const v8bf*)(gptr + (size_t)16 * NN + 16);
  v8bf cB4 = *(const v8bf*)(gptr + (size_t)32 * NN), cB5 = *(const v8bf*)(gptr + (size_t)32 * NN + 16);
  v8bf cB6 = *(const v8bf*)(gptr + (size_t)48 * NN), cB7 = *(const v8bf*)(gptr + (size_t)48 * NN + 16);

  constexpr int NITER = (NN / KSPLIT) / 32;     // 32 chunks of K=32

  #pragma unroll 4
  for (int i = 0; i < NITER - 1; ++i) {
    // issue next chunk's loads first (hidden behind current WMMAs)
    const unsigned short* ap = aptr + (i + 1) * 32;
    const unsigned short* gp = gptr + (size_t)(i + 1) * 32;
    v8bf nA0 = *(const v8bf*)(ap);
    v8bf nA1 = *(const v8bf*)(ap + 16);
    v8bf nB0 = *(const v8bf*)(gp +  0), nB1 = *(const v8bf*)(gp + 16);
    v8bf nB2 = *(const v8bf*)(gp + (size_t)16 * NN), nB3 = *(const v8bf*)(gp + (size_t)16 * NN + 16);
    v8bf nB4 = *(const v8bf*)(gp + (size_t)32 * NN), nB5 = *(const v8bf*)(gp + (size_t)32 * NN + 16);
    v8bf nB6 = *(const v8bf*)(gp + (size_t)48 * NN), nB7 = *(const v8bf*)(gp + (size_t)48 * NN + 16);

    const v16bf av = mk16(cA0, cA1);
    acc0 = __builtin_amdgcn_wmma_f32_16x16x32_bf16(false, av, false, mk16(cB0, cB1), (short)0, acc0, false, false);
    acc1 = __builtin_amdgcn_wmma_f32_16x16x32_bf16(false, av, false, mk16(cB2, cB3), (short)0, acc1, false, false);
    acc2 = __builtin_amdgcn_wmma_f32_16x16x32_bf16(false, av, false, mk16(cB4, cB5), (short)0, acc2, false, false);
    acc3 = __builtin_amdgcn_wmma_f32_16x16x32_bf16(false, av, false, mk16(cB6, cB7), (short)0, acc3, false, false);

    // rotate buffers (unrolled -> pure register renaming)
    cA0 = nA0; cA1 = nA1;
    cB0 = nB0; cB1 = nB1; cB2 = nB2; cB3 = nB3;
    cB4 = nB4; cB5 = nB5; cB6 = nB6; cB7 = nB7;
  }

  { // epilogue: last chunk
    const v16bf av = mk16(cA0, cA1);
    acc0 = __builtin_amdgcn_wmma_f32_16x16x32_bf16(false, av, false, mk16(cB0, cB1), (short)0, acc0, false, false);
    acc1 = __builtin_amdgcn_wmma_f32_16x16x32_bf16(false, av, false, mk16(cB2, cB3), (short)0, acc1, false, false);
    acc2 = __builtin_amdgcn_wmma_f32_16x16x32_bf16(false, av, false, mk16(cB4, cB5), (short)0, acc2, false, false);
    acc3 = __builtin_amdgcn_wmma_f32_16x16x32_bf16(false, av, false, mk16(cB6, cB7), (short)0, acc3, false, false);
  }

  // C/D layout: lanes 0-15 -> (M=j, N=lane); lanes 16-31 -> (M=8+j, N=lane-16)
  const int col   = lane & 15;
  const int rbase = row0 + ((lane >> 4) << 3);
  #pragma unroll
  for (int j = 0; j < 8; ++j) {
    float* o = AG + (size_t)(rbase + j) * 64;
    o[col +  0] = acc0[j];
    o[col + 16] = acc1[j];
    o[col + 32] = acc2[j];
    o[col + 48] = acc3[j];
  }
}

// ---------------- per-step: state update + output, fused with premix(t+1) ----------------
// Recurrent state flows through registers: hu/cu never round-trip through memory.
__global__ __launch_bounds__(256) void k_fused(
    const float* __restrict__ inputs,
    const float* __restrict__ hcur, const float* __restrict__ ccur,
    const float* __restrict__ AGp,
    const float* __restrict__ Whc, const float* __restrict__ Whp, const float* __restrict__ bh,
    const float* __restrict__ Wcc, const float* __restrict__ Wcp, const float* __restrict__ bc,
    const float* __restrict__ Wout, const float* __restrict__ bout,
    const float* __restrict__ Wgh, const float* __restrict__ bgh,
    const float* __restrict__ Wgc, const float* __restrict__ bgc,
    const float* __restrict__ Kz,  const float* __restrict__ Rz,
    const float* __restrict__ bz,
    unsigned short* __restrict__ G, float* __restrict__ hnext,
    float* __restrict__ cnext, float* __restrict__ out, int t, int do_premix)
{
  const int lane = threadIdx.x & 31;
  const int node = blockIdx.x * 8 + (threadIdx.x >> 5);

  const float hc = hcur[node * HH + lane];
  const float cc = ccur[node * HH + lane];

  // reduce the KSPLIT partials, then tanh
  float hg = 0.0f, cg = 0.0f;
  #pragma unroll
  for (int p = 0; p < KSPLIT; ++p) {
    hg += AGp[(size_t)p * NN * 64 + node * 64 + lane];
    cg += AGp[(size_t)p * NN * 64 + node * 64 + 32 + lane];
  }
  hg = tanhf(hg);
  cg = tanhf(cg);

  float ah = bh[lane], ac = bc[lane];
  #pragma unroll 4
  for (int k = 0; k < HH; ++k) {
    const float hck = __shfl(hc, k, 32);
    const float hgk = __shfl(hg, k, 32);
    const float cck = __shfl(cc, k, 32);
    const float cgk = __shfl(cg, k, 32);
    ah += hck * Whc[k * HH + lane] + hgk * Whp[k * HH + lane];
    ac += cck * Wcc[k * HH + lane] + cgk * Wcp[k * HH + lane];
  }
  const float hu = sigf(ah);
  const float cu = sigf(ac);

  float v = hu * Wout[lane];
  #pragma unroll
  for (int off = 16; off > 0; off >>= 1) v += __shfl_xor(v, off, 32);
  if (lane == 0) out[(size_t)node * TT + t] = v + bout[0];

  if (do_premix) {
    premix_core(hu, cu, lane, node, t + 1, inputs, Wgh, bgh, Wgc, bgc, Kz, Rz, bz,
                G, hnext, cnext);
  }
}

// ---------------- host ----------------
extern "C" void kernel_launch(void* const* d_in, const int* in_sizes, int n_in,
                              void* d_out, int out_size, void* d_ws, size_t ws_size,
                              hipStream_t stream) {
  const float* inputs = (const float*)d_in[0];
  const float* A      = (const float*)d_in[1];
  const float* Wgh    = (const float*)d_in[2];
  const float* bgh    = (const float*)d_in[3];
  const float* Wgc    = (const float*)d_in[4];
  const float* bgc    = (const float*)d_in[5];
  const float* Whc    = (const float*)d_in[6];
  const float* Whp    = (const float*)d_in[7];
  const float* bh     = (const float*)d_in[8];
  const float* Wcc    = (const float*)d_in[9];
  const float* Wcp    = (const float*)d_in[10];
  const float* bc     = (const float*)d_in[11];
  const float* Kz     = (const float*)d_in[12];
  const float* Rz     = (const float*)d_in[13];
  const float* bz     = (const float*)d_in[14];
  const float* Wout   = (const float*)d_in[15];
  const float* bout   = (const float*)d_in[16];
  float* out = (float*)d_out;

  // workspace carve-up
  unsigned char* w = (unsigned char*)d_ws;
  const size_t szA  = (size_t)NN * NN * sizeof(unsigned short);        // 32 MB
  const size_t szG  = (size_t)NN * 64 * sizeof(unsigned short);        // 512 KB
  const size_t szAG = (size_t)KSPLIT * NN * 64 * sizeof(float);        // 4 MB
  const size_t szS  = (size_t)NN * HH * sizeof(float);                 // 512 KB each
  unsigned short* Abf = (unsigned short*)(w);
  unsigned short* G   = (unsigned short*)(w + szA);
  float* AGp  = (float*)(w + szA + szG);
  float* hcur = (float*)(w + szA + szG + szAG);
  float* ccur = (float*)(w + szA + szG + szAG + szS);

  // one-time per call: A -> bf16 (stays hot in the 192MB L2 across all 365 steps)
  {
    const int n = NN * NN;
    k_f32_to_bf16<<<(n + 255) / 256, 256, 0, stream>>>(A, Abf, n);
  }

  // t = 0 pre-mix with zero initial state
  k_premix_first<<<NN / 8, 256, 0, stream>>>(inputs, Wgh, bgh, Wgc, bgc, Kz, Rz, bz,
                                             G, hcur, ccur);

  for (int t = 0; t < TT; ++t) {
    k_agemm<<<dim3(NN / 64, KSPLIT), 128, 0, stream>>>(Abf, G, AGp);
    k_fused<<<NN / 8, 256, 0, stream>>>(inputs, hcur, ccur, AGp,
                                        Whc, Whp, bh, Wcc, Wcp, bc, Wout, bout,
                                        Wgh, bgh, Wgc, bgc, Kz, Rz, bz,
                                        G, hcur, ccur, out, t, (t < TT - 1) ? 1 : 0);
  }
}
